// SelfAttention_704374637090
// MI455X (gfx1250) — compile-verified, matrix operand order checked by reading, not run on hardware
//
#include <hip/hip_runtime.h>
#include <hip/hip_bf16.h>
#include <stdint.h>

// Self-attention, B=8 S=1024 E=1024 H=16 d=64.
// ~103 GFLOP, ~150MB mandatory HBM traffic -> compute bound on WMMA pipes.
// fp32 -> bf16 at staging, f32 accumulate in v_wmma_f32_16x16x32_bf16,
// flash-attention streaming. Tile staging uses TDM (tensor_load_to_lds) and
// async global->LDS copies where the copy is conversion/transpose-free,
// guarded by __has_builtin with manual fallbacks.

#define Bsz 8
#define Ssz 1024
#define Esz 1024
#define Hn  16
#define Dh  64

typedef __attribute__((ext_vector_type(16))) __bf16 bf16x16;
typedef __attribute__((ext_vector_type(8)))  float  f32x8;
typedef __attribute__((ext_vector_type(4)))  int    v4i;

#if defined(__has_builtin)
#  if __has_builtin(__builtin_amdgcn_global_load_async_to_lds_b128) && \
      __has_builtin(__builtin_amdgcn_s_wait_asynccnt)
#    define HAVE_ASYNC_LDS 1
#  endif
#  if __has_builtin(__builtin_amdgcn_tensor_load_to_lds) && \
      __has_builtin(__builtin_amdgcn_s_wait_tensorcnt) && \
      !__has_include(<hip/amd_detail/amd_gfx1250_TDM.h>)
#    define HAVE_TDM5 1   /* ROCm 7.2 5-arg tensor_load_to_lds */
#  endif
#endif
#ifndef HAVE_ASYNC_LDS
#define HAVE_ASYNC_LDS 0
#endif
#ifndef HAVE_TDM5
#define HAVE_TDM5 0
#endif

#if HAVE_TDM5
typedef __attribute__((ext_vector_type(4))) unsigned int u32x4;
typedef __attribute__((ext_vector_type(8))) int          i32x8;
typedef __attribute__((ext_vector_type(4))) int          i32x4;
#endif

// async-LDS b128 builtin takes v4i32 pointers: AS1 (global) src, AS3 (LDS) dst
#define GV4(p) ((__attribute__((address_space(1))) v4i*)(p))
#define LV4(p) ((__attribute__((address_space(3))) v4i*)(p))

union FragU {
  uint4 q[2];
  bf16x16 v;
};

__device__ __forceinline__ unsigned short f2bf(float f) {
  unsigned int u = __builtin_bit_cast(unsigned int, f);
  u += 0x7fffu + ((u >> 16) & 1u);   // round-to-nearest-even
  return (unsigned short)(u >> 16);
}

// 16-bit A/B fragment load per CDNA5 ISA 7.12.2:
// lane (m = lane%16, h = lane/16) holds row m, K = {8h..8h+7, 8h+16..8h+23}
// -> two contiguous 16B chunks when rows are contiguous in LDS/memory.
__device__ __forceinline__ bf16x16 load_frag(const unsigned short* base,
                                             int stride_u16, int m, int h) {
  FragU f;
  const unsigned short* p = base + m * stride_u16 + 8 * h;
  f.q[0] = *(const uint4*)(p);
  f.q[1] = *(const uint4*)(p + 16);
  return f.v;
}

__device__ __forceinline__ f32x8 wmma_bf16(bf16x16 a, bf16x16 b, f32x8 c) {
  return __builtin_amdgcn_wmma_f32_16x16x32_bf16(false, a, false, b,
                                                 (short)0, c, false, false);
}

// ---------------------------------------------------------------------------
// GEMM: C[M,N] = A[M,K] * B[K,N].  A fp32 or bf16, B fp32, C bf16 or fp32.
// Block tile 128x128, BK=32, 8 waves (256 threads), each wave 32x64.
// When A is already bf16, the A tile is staged with per-lane async
// global->LDS b128 copies (ASYNCcnt path); otherwise staged through VGPRs
// with fused fp32->bf16 conversion.
// ---------------------------------------------------------------------------
template<bool A_BF16, bool C_BF16>
__global__ __launch_bounds__(256) void gemm_kernel(const void* __restrict__ Ap,
                                                   const float* __restrict__ Bp,
                                                   void* __restrict__ Cp,
                                                   int M, int N, int K,
                                                   int lda, int ldb, int ldc) {
  __shared__ unsigned short As[128 * 40];   // A tile, row-major, pad 40
  __shared__ unsigned short Bt[128 * 40];   // B tile transposed: Bt[n][k]

  const int tid  = threadIdx.x;
  const int lane = tid & 31;
  const int w    = tid >> 5;
  const int n_   = lane & 15;
  const int hl   = lane >> 4;

  const int m0 = blockIdx.y * 128;
  const int n0 = blockIdx.x * 128;
  const int mw = (w & 3) * 32;   // wave M offset
  const int nw = (w >> 2) * 64;  // wave N offset

  f32x8 acc[2][4];
#pragma unroll
  for (int i = 0; i < 2; ++i)
#pragma unroll
    for (int j = 0; j < 4; ++j)
#pragma unroll
      for (int r = 0; r < 8; ++r) acc[i][j][r] = 0.0f;

  const int ksteps = K / 32;
  for (int kt = 0; kt < ksteps; ++kt) {
    // ---- stage A tile 128x32 ----
#if HAVE_ASYNC_LDS
    if (A_BF16) {
      // pure bf16 copy: per-lane async global->LDS, 16B per lane per op
      int row = tid >> 2;          // 0..63
      int c8  = (tid & 3) * 8;     // ushort col offset 0/8/16/24
#pragma unroll
      for (int i = 0; i < 2; ++i) {
        int r2 = row + 64 * i;
        const unsigned short* src =
            (const unsigned short*)Ap + (size_t)(m0 + r2) * lda + kt * 32 + c8;
        __builtin_amdgcn_global_load_async_to_lds_b128(
            GV4(src), LV4(&As[r2 * 40 + c8]), 0, 0);
      }
    } else
#endif
    {
      int row = tid >> 3;
      int col = (tid & 7) * 4;
#pragma unroll
      for (int i = 0; i < 4; ++i, row += 32) {
        if (A_BF16) {
          const unsigned short* src =
              (const unsigned short*)Ap + (size_t)(m0 + row) * lda + kt * 32 + col;
          *(uint2*)&As[row * 40 + col] = *(const uint2*)src;
        } else {
          const float* src = (const float*)Ap + (size_t)(m0 + row) * lda + kt * 32 + col;
          float4 v = *(const float4*)src;
          unsigned short* d = &As[row * 40 + col];
          d[0] = f2bf(v.x); d[1] = f2bf(v.y); d[2] = f2bf(v.z); d[3] = f2bf(v.w);
        }
      }
    }
    // ---- stage B tile 32x128 transposed into Bt[n][k] (fp32->bf16) ----
    {
      int kk = tid >> 5;
      int nn = (tid & 31) * 4;
#pragma unroll
      for (int i = 0; i < 4; ++i, kk += 8) {
        const float* src = Bp + (size_t)(kt * 32 + kk) * ldb + n0 + nn;
        float4 v = *(const float4*)src;
        Bt[(nn + 0) * 40 + kk] = f2bf(v.x);
        Bt[(nn + 1) * 40 + kk] = f2bf(v.y);
        Bt[(nn + 2) * 40 + kk] = f2bf(v.z);
        Bt[(nn + 3) * 40 + kk] = f2bf(v.w);
      }
    }
    // L2 prefetch of next K tiles (global_prefetch_b8)
    if (kt + 1 < ksteps) {
      if (A_BF16)
        __builtin_prefetch((const unsigned short*)Ap +
                           (size_t)(m0 + (tid >> 1)) * lda + (kt + 1) * 32, 0, 1);
      else
        __builtin_prefetch((const float*)Ap +
                           (size_t)(m0 + (tid >> 1)) * lda + (kt + 1) * 32, 0, 1);
      __builtin_prefetch(Bp + (size_t)((kt + 1) * 32 + (tid & 31)) * ldb + n0, 0, 1);
    }
#if HAVE_ASYNC_LDS
    if (A_BF16) __builtin_amdgcn_s_wait_asynccnt(0);
#endif
    __syncthreads();

    bf16x16 af[2], bfr[4];
#pragma unroll
    for (int mf = 0; mf < 2; ++mf)
      af[mf] = load_frag(&As[(mw + mf * 16) * 40], 40, n_, hl);
#pragma unroll
    for (int nf = 0; nf < 4; ++nf)
      bfr[nf] = load_frag(&Bt[(nw + nf * 16) * 40], 40, n_, hl);
#pragma unroll
    for (int mf = 0; mf < 2; ++mf)
#pragma unroll
      for (int nf = 0; nf < 4; ++nf)
        acc[mf][nf] = wmma_bf16(af[mf], bfr[nf], acc[mf][nf]);

    __syncthreads();
  }

  // ---- epilogue: C layout VGPR r -> row r + 8*hl, lane n_ -> col ----
#pragma unroll
  for (int mf = 0; mf < 2; ++mf)
#pragma unroll
    for (int nf = 0; nf < 4; ++nf)
#pragma unroll
      for (int r = 0; r < 8; ++r) {
        int row = m0 + mw + mf * 16 + r + 8 * hl;
        int col = n0 + nw + nf * 16 + n_;
        float v = acc[mf][nf][r];
        if (C_BF16) ((unsigned short*)Cp)[(size_t)row * ldc + col] = f2bf(v);
        else        ((float*)Cp)[(size_t)row * ldc + col] = v;
      }
}

// ---------------------------------------------------------------------------
// Causal flash attention.  One block = 4 waves (128 threads) handles one
// (batch, head, 64 query rows).  Each wave owns 16 q rows.
// qkv: bf16 [B*S, 3E] (q | k | v).  y: bf16 [B*S, E].
// K tile staged via TDM 2-D descriptor (row 128B, global stride 6144B,
// LDS pad 16B/row expressed as pad_interval=32 dwords / pad_amount=4 dwords).
// ---------------------------------------------------------------------------
__global__ __launch_bounds__(128) void attn_kernel(const unsigned short* __restrict__ qkv,
                                                   unsigned short* __restrict__ y) {
  __shared__ unsigned short Kt[64 * 72];        // K tile row-major  [krow][d]
  __shared__ unsigned short Vt[64 * 72];        // V tile transposed [d][krow]
  __shared__ unsigned short Pb[4 * 16 * 72];    // per-wave P relayout buffer

  const int tid  = threadIdx.x;
  const int lane = tid & 31;
  const int w    = tid >> 5;       // 0..3
  const int n_   = lane & 15;
  const int hl   = lane >> 4;

  const int bh = blockIdx.x;
  const int b  = bh / Hn;
  const int h  = bh % Hn;
  const int qb = blockIdx.y;       // 64-row query block
  const int q0 = qb * 64;

  const int ld = 3 * Esz;          // qkv row stride (bf16 elements)
  const size_t rowbase = (size_t)b * Ssz;

  // Q fragments (persistent): rows q0+16w .. +15, d split into two 32-chunks
  bf16x16 qf[2];
  {
    const unsigned short* qp = qkv + (rowbase + q0 + w * 16) * ld + h * Dh;
#pragma unroll
    for (int f = 0; f < 2; ++f)
      qf[f] = load_frag(qp + 32 * f, ld, n_, hl);
  }

  float m_r[8], l_r[8];
  f32x8 o[4];
#pragma unroll
  for (int r = 0; r < 8; ++r) { m_r[r] = -3.0e38f; l_r[r] = 0.0f; }
#pragma unroll
  for (int nf = 0; nf < 4; ++nf)
#pragma unroll
    for (int r = 0; r < 8; ++r) o[nf][r] = 0.0f;

  const float scale = 0.125f;      // 1/sqrt(64)

  for (int kt = 0; kt <= qb; ++kt) {
    // ---- stage K tile ----
#if HAVE_TDM5
    if (w == 0) {
      const unsigned short* ks = qkv + (rowbase + kt * 64) * ld + Esz + h * Dh;
      unsigned long long ga = (unsigned long long)(uintptr_t)ks;
      unsigned lds_off = (unsigned)(unsigned long long)
          (__attribute__((address_space(3))) unsigned short*)Kt;
      u32x4 g0;
      g0[0] = 1u;                                   // count=1
      g0[1] = lds_off;                              // lds_addr (bytes)
      g0[2] = (unsigned)ga;                         // global_addr[31:0]
      g0[3] = (unsigned)((ga >> 32) & 0x01FFFFFFu)  // global_addr[56:32]
              | (2u << 30);                         // type = image
      i32x8 g1;
      g1[0] = (1 << 16)        // data_size = 2 bytes
            | (1 << 20)        // pad_enable
            | (4 << 22)        // pad_interval: 32 dwords (=128B, one row)
            | (3 << 25);       // pad_amount:   4 dwords (=16B)
      g1[1] = (3072 << 16);    // tensor_dim0 = 3072 (low16); barrier addr 0
      g1[2] = 0;               // tensor_dim0 hi16 = 0; tensor_dim1 low16 = 0
      g1[3] = 1 | (64 << 16);  // tensor_dim1 hi16 = 1 (dim1=65536); tile_dim0=64
      g1[4] = 64;              // tile_dim1 = 64; tile_dim2 = 0
      g1[5] = 3072;            // tensor_dim0_stride low32
      g1[6] = 0;               // stride hi16, tensor_dim1_stride low16
      g1[7] = 0;
      i32x4 z4 = {0, 0, 0, 0};
      __builtin_amdgcn_tensor_load_to_lds(g0, g1, z4, z4, 0);
      __builtin_amdgcn_s_wait_tensorcnt(0);
    }
#else
    {
      int row = tid >> 1;
      int cb  = (tid & 1) * 32;
      const unsigned short* ks =
          qkv + (rowbase + kt * 64 + row) * ld + Esz + h * Dh + cb;
      *(uint4*)&Kt[row * 72 + cb +  0] = *(const uint4*)(ks +  0);
      *(uint4*)&Kt[row * 72 + cb +  8] = *(const uint4*)(ks +  8);
      *(uint4*)&Kt[row * 72 + cb + 16] = *(const uint4*)(ks + 16);
      *(uint4*)&Kt[row * 72 + cb + 24] = *(const uint4*)(ks + 24);
    }
#endif
    // ---- stage V tile transposed: Vt[d][k] ----
    {
      int d4 = (tid & 15) * 4;
#pragma unroll
      for (int j = 0; j < 8; ++j) {
        int k = (tid >> 4) + 8 * j;
        const unsigned short* vs =
            qkv + (rowbase + kt * 64 + k) * ld + 2 * Esz + h * Dh + d4;
        uint2 v = *(const uint2*)vs;
        Vt[(d4 + 0) * 72 + k] = (unsigned short)(v.x & 0xffffu);
        Vt[(d4 + 1) * 72 + k] = (unsigned short)(v.x >> 16);
        Vt[(d4 + 2) * 72 + k] = (unsigned short)(v.y & 0xffffu);
        Vt[(d4 + 3) * 72 + k] = (unsigned short)(v.y >> 16);
      }
    }
    __syncthreads();

    // ---- scores S = Q K^T (16 q rows x 64 k cols) ----
    f32x8 sc[4];
#pragma unroll
    for (int kc = 0; kc < 4; ++kc) {
#pragma unroll
      for (int r = 0; r < 8; ++r) sc[kc][r] = 0.0f;
#pragma unroll
      for (int f = 0; f < 2; ++f) {
        bf16x16 kf = load_frag(&Kt[(kc * 16) * 72 + 32 * f], 72, n_, hl);
        sc[kc] = wmma_bf16(qf[f], kf, sc[kc]);
      }
    }

    // scale + causal mask (diagonal block only)
#pragma unroll
    for (int kc = 0; kc < 4; ++kc)
#pragma unroll
      for (int r = 0; r < 8; ++r) sc[kc][r] *= scale;
    if (kt == qb) {
#pragma unroll
      for (int kc = 0; kc < 4; ++kc)
#pragma unroll
        for (int r = 0; r < 8; ++r) {
          int qrel = w * 16 + r + 8 * hl;
          int krel = kc * 16 + n_;
          if (krel > qrel) sc[kc][r] = -3.0e38f;
        }
    }

    // ---- online softmax ----
    float mx[8];
#pragma unroll
    for (int r = 0; r < 8; ++r) {
      mx[r] = sc[0][r];
#pragma unroll
      for (int kc = 1; kc < 4; ++kc) mx[r] = fmaxf(mx[r], sc[kc][r]);
    }
#pragma unroll
    for (int off = 1; off < 16; off <<= 1)
#pragma unroll
      for (int r = 0; r < 8; ++r)
        mx[r] = fmaxf(mx[r], __shfl_xor(mx[r], off, 32));

    float al[8];
#pragma unroll
    for (int r = 0; r < 8; ++r) {
      float mnew = fmaxf(m_r[r], mx[r]);
      al[r] = __expf(m_r[r] - mnew);
      m_r[r] = mnew;
      l_r[r] *= al[r];
    }
#pragma unroll
    for (int nf = 0; nf < 4; ++nf)
#pragma unroll
      for (int r = 0; r < 8; ++r) o[nf][r] *= al[r];

    float rs[8];
#pragma unroll
    for (int r = 0; r < 8; ++r) rs[r] = 0.0f;
#pragma unroll
    for (int kc = 0; kc < 4; ++kc)
#pragma unroll
      for (int r = 0; r < 8; ++r) {
        float p = __expf(sc[kc][r] - m_r[r]);
        sc[kc][r] = p;
        rs[r] += p;
      }
#pragma unroll
    for (int off = 1; off < 16; off <<= 1)
#pragma unroll
      for (int r = 0; r < 8; ++r) rs[r] += __shfl_xor(rs[r], off, 32);
#pragma unroll
    for (int r = 0; r < 8; ++r) l_r[r] += rs[r];

    // ---- P -> LDS (C layout -> A-fragment layout round trip) ----
    unsigned short* pw = &Pb[w * 16 * 72];
#pragma unroll
    for (int kc = 0; kc < 4; ++kc)
#pragma unroll
      for (int r = 0; r < 8; ++r)
        pw[(r + 8 * hl) * 72 + kc * 16 + n_] = f2bf(sc[kc][r]);

    // ---- O += P V ----
#pragma unroll
    for (int kc2 = 0; kc2 < 2; ++kc2) {
      bf16x16 pa = load_frag(&Pb[w * 16 * 72 + 32 * kc2], 72, n_, hl);
#pragma unroll
      for (int nf = 0; nf < 4; ++nf) {
        bf16x16 vb = load_frag(&Vt[(nf * 16) * 72 + 32 * kc2], 72, n_, hl);
        o[nf] = wmma_bf16(pa, vb, o[nf]);
      }
    }
    __syncthreads();
  }

  // ---- epilogue: normalize and store bf16 y[b, q, h*64 + d] ----
#pragma unroll
  for (int r = 0; r < 8; ++r) {
    float inv = 1.0f / l_r[r];
    int qg = q0 + w * 16 + r + 8 * hl;
#pragma unroll
    for (int nf = 0; nf < 4; ++nf) {
      int d = nf * 16 + n_;
      y[(rowbase + qg) * Esz + h * Dh + d] = f2bf(o[nf][r] * inv);
    }
  }
}

// ---------------------------------------------------------------------------
extern "C" void kernel_launch(void* const* d_in, const int* in_sizes, int n_in,
                              void* d_out, int out_size, void* d_ws, size_t ws_size,
                              hipStream_t stream) {
  const float* x  = (const float*)d_in[0];   // [8,1024,1024]
  const float* Wa = (const float*)d_in[1];   // [1024,3072]
  const float* Wp = (const float*)d_in[2];   // [1024,1024]
  float* out = (float*)d_out;                // [8,1024,1024]

  unsigned short* qkv = (unsigned short*)d_ws;                    // 8192x3072 bf16
  unsigned short* y   = qkv + (size_t)Bsz * Ssz * 3 * Esz;        // 8192x1024 bf16

  const int M = Bsz * Ssz;   // 8192

  // 1) qkv = x @ W_attn   (fp32 A, bf16 C)
  dim3 g1(3 * Esz / 128, M / 128);
  gemm_kernel<false, true><<<g1, dim3(256), 0, stream>>>(
      x, Wa, qkv, M, 3 * Esz, Esz, Esz, 3 * Esz, 3 * Esz);

  // 2) flash attention (causal)
  dim3 ga(Bsz * Hn, Ssz / 64);
  attn_kernel<<<ga, dim3(128), 0, stream>>>(qkv, y);

  // 3) out = y @ W_proj   (bf16 A, fp32 C)
  dim3 g2(Esz / 128, M / 128);
  gemm_kernel<true, false><<<g2, dim3(256), 0, stream>>>(
      y, Wp, out, M, Esz, Esz, Esz, Esz, Esz);
}
